// SRWAB_81484119540016
// MI455X (gfx1250) — compile-verified, matrix operand order checked by reading, not run on hardware
//
#include <hip/hip_runtime.h>
#include <cstdint>
#include <cstddef>

// ---------------------------------------------------------------------------
// Problem constants
// ---------------------------------------------------------------------------
constexpr int kB   = 4;
constexpr int kC   = 384;
constexpr int kH   = 161;
constexpr int kW   = 161;
constexpr int kHW  = kH * kW;          // 25921
constexpr int kNH  = 8;
constexpr int kHD  = 48;
constexpr int kNwinH = 41;             // ceil(161/4)
constexpr int kNwinW = 11;             // ceil(161/16)
constexpr int kNWin  = kB * kNwinH * kNwinW;   // 1804 windows
constexpr int kNT    = kNWin * 64;             // 115456 window tokens
constexpr int kTPI   = (kHW + 15) / 16;        // 1621 16-token tiles / image

typedef __attribute__((ext_vector_type(16))) _Float16 v16h;
typedef __attribute__((ext_vector_type(8)))  float    v8f;
typedef unsigned int u32x4 __attribute__((ext_vector_type(4)));
typedef int          i32x4 __attribute__((ext_vector_type(4)));
typedef int          i32x8 __attribute__((ext_vector_type(8)));

// Tensor Data Mover availability (device pass only; host pass uses fallback).
#if defined(__HIP_DEVICE_COMPILE__) && defined(__gfx1250__) && \
    __has_builtin(__builtin_amdgcn_tensor_load_to_lds)
#define ATHENA_USE_TDM 1
#else
#define ATHENA_USE_TDM 0
#endif

union FragH {
  v16h v;
  _Float16 h[16];
  uint4 q[2];
};

// Load one wave-lane's share of a 16x32 f16 WMMA operand.
// Per CDNA5 ISA: lane L(0-15) holds row/col L, K chunks {0..7} and {16..23};
// lane L+16 holds K chunks {8..15} and {24..31}.  Caller passes p pointing at
// the lane's row base + k0 + (lane>=16 ? 8 : 0); both 16-byte chunks are
// contiguous in memory for row-major (K-contiguous) storage.
__device__ __forceinline__ v16h frag16(const _Float16* p) {
  FragH f;
  f.q[0] = *(const uint4*)(p);
  f.q[1] = *(const uint4*)(p + 16);
  return f.v;
}

__device__ __forceinline__ v8f wmma_f16(v16h a, v16h b, v8f c) {
  return __builtin_amdgcn_wmma_f32_16x16x32_f16(false, a, false, b, (short)0, c,
                                                false, false);
}

#if ATHENA_USE_TDM
// TDM: DMA a 2D f16 tile (rows x cols, global row stride rowStride elements)
// from global memory into LDS at byte offset ldsOff.  Descriptor layout per
// CDNA5 ISA (D# groups 0/1; groups 2/3 zero for 2D).
__device__ __forceinline__ void tdm_load_tile_f16(uint32_t ldsOff,
                                                  const void* gptr, int cols,
                                                  int rows, int rowStride) {
  uint64_t ga = (uint64_t)(uintptr_t)gptr;
  u32x4 g0;
  g0[0] = 1u;                                   // count=1, user descriptor
  g0[1] = ldsOff;                               // lds_addr (bytes)
  g0[2] = (uint32_t)ga;                         // global_addr[31:0]
  g0[3] = (uint32_t)((ga >> 32) & 0x01FFFFFFu)  // global_addr[56:32]
          | (2u << 30);                         // type = 2 ("image")
  i32x8 g1;
  g1[0] = (1 << 16);                            // workgroup_mask=0, data_size=2B
  g1[1] = (int)(((uint32_t)cols & 0xFFFFu) << 16);          // tensor_dim0 lo16
  g1[2] = (int)((((uint32_t)cols >> 16) & 0xFFFFu)          // tensor_dim0 hi16
                | (((uint32_t)rows & 0xFFFFu) << 16));      // tensor_dim1 lo16
  g1[3] = (int)((((uint32_t)rows >> 16) & 0xFFFFu)          // tensor_dim1 hi16
                | (((uint32_t)cols & 0xFFFFu) << 16));      // tile_dim0
  g1[4] = (int)((uint32_t)rows & 0xFFFFu);                  // tile_dim1 (dim2=0)
  g1[5] = rowStride;                                        // dim0_stride lo32
  g1[6] = 0;                                                // stride hi / dim1_stride
  g1[7] = 0;
  i32x4 z4 = {0, 0, 0, 0};
#if __has_include(<hip/amd_detail/amd_gfx1250_TDM.h>)
  i32x8 z8 = {0, 0, 0, 0, 0, 0, 0, 0};
  __builtin_amdgcn_tensor_load_to_lds(g0, g1, z4, z4, z8, 0);   // clang-23 form
#else
  __builtin_amdgcn_tensor_load_to_lds(g0, g1, z4, z4, 0);       // ROCm 7.2 form
#endif
}
#endif  // ATHENA_USE_TDM

// ---------------------------------------------------------------------------
// 1) Fold inference BN into weight rows: W'[r,k] = W[r,k]*s(r) (f16),
//    bias'[r] = b[r] - s(r)*rm[r], with s = g*rsqrt(rv+eps_bn).
//    bn is jnp.stack([g,b,rm,rv]) -> 4 contiguous length-R arrays.
// ---------------------------------------------------------------------------
__global__ void fold_bn_kernel(const float* __restrict__ Wsrc,
                               const float* __restrict__ bn,
                               int R, int K,
                               _Float16* __restrict__ Wo,
                               float* __restrict__ Bo) {
  int i = blockIdx.x * 256 + threadIdx.x;
  if (i < R * K) {
    int r = i / K;
    float s = bn[r] * rsqrtf(bn[3 * R + r] + 1e-3f);
    Wo[i] = (_Float16)(Wsrc[i] * s);
  }
  if (i < R) {
    float s = bn[i] * rsqrtf(bn[3 * R + i] + 1e-3f);
    Bo[i] = bn[R + i] - s * bn[2 * R + i];
  }
}

// ---------------------------------------------------------------------------
// 2) NCHW -> NHWC transpose + LayerNorm-1 + window partition.
//    Writes xh (f32 residual, (B,HW,C)) and winX (f16, (N,64,C), LN'ed).
//    Pad rows of winX are pre-zeroed by a memset (reference pads AFTER LN
//    with zeros and does not mask attention).
// ---------------------------------------------------------------------------
__global__ __launch_bounds__(256) void ln1_window_kernel(
    const float* __restrict__ x, const float* __restrict__ g,
    const float* __restrict__ b, float* __restrict__ xh,
    _Float16* __restrict__ winX) {
  __shared__ float tile[16 * 385];
  int bi = blockIdx.x / kTPI;
  int t0 = (blockIdx.x % kTPI) * 16;

  // Load 384 x 16 tile: lanes run along hw (coalesced in x's W dim).
  for (int idx = threadIdx.x; idx < kC * 16; idx += 256) {
    int c = idx >> 4, tok = idx & 15;
    int hw = t0 + tok;
    float v = 0.f;
    if (hw < kHW) v = x[((size_t)bi * kC + c) * kHW + hw];
    tile[tok * 385 + c] = v;
  }
  __syncthreads();

  int tok = threadIdx.x >> 4, j = threadIdx.x & 15;  // 16 lanes per token
  float sum = 0.f, sq = 0.f;
  for (int c = j; c < kC; c += 16) {
    float v = tile[tok * 385 + c];
    sum += v; sq += v * v;
  }
  for (int d = 8; d >= 1; d >>= 1) {
    sum += __shfl_xor(sum, d);
    sq  += __shfl_xor(sq, d);
  }
  float mean = sum * (1.f / kC);
  float var  = sq * (1.f / kC) - mean * mean;
  float inv  = rsqrtf(var + 1e-5f);

  int hw = t0 + tok;
  if (hw < kHW) {
    int h = hw / kW, w = hw % kW;
    int n = (bi * kNwinH + (h >> 2)) * kNwinW + (w >> 4);
    int l = ((h & 3) << 4) | (w & 15);
    size_t xb = ((size_t)bi * kHW + hw) * kC;
    size_t wb = ((size_t)n * 64 + l) * kC;
    for (int c = j; c < kC; c += 16) {
      float v = tile[tok * 385 + c];
      xh[xb + c] = v;
      winX[wb + c] = (_Float16)((v - mean) * inv * g[c] + b[c]);
    }
  }
}

// ---------------------------------------------------------------------------
// 3) Window GEMM:  Out[64 x COLS] = A_window[64 x K] @ W'[COLS x K]^T + bias.
//    One workgroup (8 waves) per window; 48KB LDS A tile staged by the
//    Tensor Data Mover (one wave issues tensor_load_to_lds, waits on
//    TENSORcnt, barrier publishes).  B fragments come straight from the
//    (L2-resident) folded weights.
//    MODE 0: scatter to Q/K/V (N,NH,64,48) f16        (qkv, COLS=1152)
//    MODE 1: residual add into xh at real tokens      (proj, COLS=384)
//    MODE 2: f16 row-major out, optional SiLU         (mlp1, COLS=768)
//    MODE 3: f32 row-major out                        (mlp2, COLS=384, K=768)
// ---------------------------------------------------------------------------
template <int KDIM, int COLS, int MODE, bool SILU>
__global__ __launch_bounds__(256) void win_gemm_kernel(
    const _Float16* __restrict__ A, const _Float16* __restrict__ Wt,
    const float* __restrict__ bias, _Float16* __restrict__ outH,
    float* __restrict__ outF, _Float16* __restrict__ Qo,
    _Float16* __restrict__ Ko, _Float16* __restrict__ Vo) {
  __shared__ __align__(16) _Float16 As[64 * 384];
  const int n = blockIdx.x;
  const _Float16* Ag = A + (size_t)n * 64 * KDIM;
  const int wv = threadIdx.x >> 5, lane = threadIdx.x & 31;
  const int rowIn = lane & 15;
  const int kHalf = (lane >> 4) * 8;
  const int rhi   = (lane >> 4) * 8;
  constexpr int NPASS  = COLS / 128;   // 8 waves x 16 cols per pass
  constexpr int CHUNKS = KDIM / 384;

  for (int pass = 0; pass < NPASS; ++pass) {
    const int ct   = pass * 8 + wv;
    const int colG = ct * 16 + rowIn;
    const _Float16* Wrow = Wt + (size_t)colG * KDIM;
    __builtin_prefetch(Wrow, 0, 0);

    v8f acc[4];
    for (int r = 0; r < 4; ++r)
      for (int e = 0; e < 8; ++e) acc[r][e] = 0.f;

    for (int kc = 0; kc < KDIM; kc += 384) {
      if (CHUNKS > 1 || pass == 0) {        // uniform across block
        __syncthreads();
#if ATHENA_USE_TDM
        if (wv == 0) {
          // 64 x 384 f16 tile, global row stride KDIM elements.
          tdm_load_tile_f16((uint32_t)(uintptr_t)(void*)As, Ag + kc, 384, 64,
                            KDIM);
          __builtin_amdgcn_s_wait_tensorcnt(0);
        }
#else
        for (int i = threadIdx.x * 8; i < 64 * 384; i += 256 * 8) {
          int r = i / 384, c = i - r * 384;
          *(uint4*)(As + i) = *(const uint4*)(Ag + (size_t)r * KDIM + kc + c);
        }
#endif
        __syncthreads();
      }
      for (int k0 = 0; k0 < 384; k0 += 32) {
        v16h bfr = frag16(Wrow + kc + k0 + kHalf);
#pragma unroll
        for (int rt = 0; rt < 4; ++rt) {
          v16h afr = frag16(As + (rt * 16 + rowIn) * 384 + k0 + kHalf);
          acc[rt] = wmma_f16(afr, bfr, acc[rt]);
        }
      }
    }

    const float bv = bias[colG];
#pragma unroll
    for (int rt = 0; rt < 4; ++rt) {
#pragma unroll
      for (int v = 0; v < 8; ++v) {
        int row = rt * 16 + v + rhi;        // C-matrix lane layout
        float val = acc[rt][v] + bv;
        if constexpr (SILU) val = val / (1.f + __expf(-val));
        if constexpr (MODE == 0) {
          int sec = colG / 384;
          int rem = colG - sec * 384;
          int hh = rem / 48, d = rem - hh * 48;
          _Float16* P = (sec == 0) ? Qo : ((sec == 1) ? Ko : Vo);
          P[(((size_t)n * 8 + hh) * 64 + row) * 48 + d] = (_Float16)val;
        } else if constexpr (MODE == 1) {
          int bi = n / (kNwinH * kNwinW);
          int rem = n - bi * (kNwinH * kNwinW);
          int wh = rem / kNwinW, ww = rem - wh * kNwinW;
          int h = wh * 4 + (row >> 4);
          int w = ww * 16 + (row & 15);
          if (h < kH && w < kW)
            outF[((size_t)bi * kHW + h * kW + w) * kC + colG] += val;
        } else if constexpr (MODE == 2) {
          outH[((size_t)n * 64 + row) * COLS + colG] = (_Float16)val;
        } else {
          outF[((size_t)n * 64 + row) * COLS + colG] = val;
        }
      }
    }
  }
}

// ---------------------------------------------------------------------------
// 4) Attention: one block (4 waves) per (window, head).
//    S = Q K^T * scale (HD=48 zero-padded to K=64), softmax over keys,
//    O = P V.  P passes through LDS to get A-operand fragment layout.
// ---------------------------------------------------------------------------
__global__ __launch_bounds__(128) void attn_kernel(
    const _Float16* __restrict__ Q, const _Float16* __restrict__ K,
    const _Float16* __restrict__ V, _Float16* __restrict__ O) {
  __shared__ __align__(16) _Float16 Qs[64 * 64];
  __shared__ __align__(16) _Float16 Ks[64 * 64];
  __shared__ __align__(16) _Float16 Vt[64 * 64];   // transposed: Vt[d][l]
  __shared__ __align__(16) _Float16 Ps[64 * 64];
  const int n  = blockIdx.x >> 3;
  const int hh = blockIdx.x & 7;
  const size_t base = ((size_t)n * 8 + hh) * 64 * 48;
  const _Float16* Qg = Q + base;
  const _Float16* Kg = K + base;
  const _Float16* Vg = V + base;

  for (int i = threadIdx.x; i < 64 * 48; i += 128) {
    int l = i / 48, d = i - l * 48;
    Qs[l * 64 + d] = Qg[i];
    Ks[l * 64 + d] = Kg[i];
    Vt[d * 64 + l] = Vg[i];
  }
  for (int i = threadIdx.x; i < 64 * 16; i += 128) {  // zero-pad K-dim 48..63
    int l = i >> 4, d = 48 + (i & 15);
    Qs[l * 64 + d] = (_Float16)0.f;
    Ks[l * 64 + d] = (_Float16)0.f;
  }
  __syncthreads();

  const int wv = threadIdx.x >> 5, lane = threadIdx.x & 31;
  const int rowIn = lane & 15;
  const int kHalf = (lane >> 4) * 8;
  const int rhi   = (lane >> 4) * 8;

  // S = Q K^T : wave wv owns query rows 16*wv..16*wv+15, all 64 key cols.
  v8f acc[4];
  for (int j = 0; j < 4; ++j)
    for (int e = 0; e < 8; ++e) acc[j][e] = 0.f;
  for (int k0 = 0; k0 < 64; k0 += 32) {
    v16h a = frag16(Qs + (16 * wv + rowIn) * 64 + k0 + kHalf);
#pragma unroll
    for (int j = 0; j < 4; ++j) {
      v16h b = frag16(Ks + (j * 16 + rowIn) * 64 + k0 + kHalf);
      acc[j] = wmma_f16(a, b, acc[j]);
    }
  }

  // softmax over 64 keys; a row lives in one VGPR index across 16 lanes.
  const float scale = 0.14433756729740643f;  // 1/sqrt(48)
#pragma unroll
  for (int v = 0; v < 8; ++v) {
    float mx = -3.4e38f;
    for (int j = 0; j < 4; ++j) {
      acc[j][v] *= scale;
      mx = fmaxf(mx, acc[j][v]);
    }
    for (int d = 8; d >= 1; d >>= 1) mx = fmaxf(mx, __shfl_xor(mx, d));
    float s = 0.f;
    for (int j = 0; j < 4; ++j) {
      float e = __expf(acc[j][v] - mx);
      acc[j][v] = e;
      s += e;
    }
    for (int d = 8; d >= 1; d >>= 1) s += __shfl_xor(s, d);
    float inv = 1.f / s;
    int row = 16 * wv + v + rhi;
    for (int j = 0; j < 4; ++j)
      Ps[row * 64 + j * 16 + rowIn] = (_Float16)(acc[j][v] * inv);
  }
  __syncthreads();

  // O = P V : cols d = 0..47, K-dim = 64 keys.
  v8f oc[3];
  for (int c = 0; c < 3; ++c)
    for (int e = 0; e < 8; ++e) oc[c][e] = 0.f;
  for (int k0 = 0; k0 < 64; k0 += 32) {
    v16h a = frag16(Ps + (16 * wv + rowIn) * 64 + k0 + kHalf);
#pragma unroll
    for (int ct = 0; ct < 3; ++ct) {
      v16h b = frag16(Vt + (ct * 16 + rowIn) * 64 + k0 + kHalf);
      oc[ct] = wmma_f16(a, b, oc[ct]);
    }
  }
#pragma unroll
  for (int ct = 0; ct < 3; ++ct)
#pragma unroll
    for (int v = 0; v < 8; ++v) {
      int l = 16 * wv + v + rhi;
      int d = ct * 16 + rowIn;
      O[((size_t)n * 64 + l) * kC + hh * 48 + d] = (_Float16)oc[ct][v];
    }
}

// ---------------------------------------------------------------------------
// 5) LayerNorm-2 on residual stream -> window-layout f16 (pad rows unused).
// ---------------------------------------------------------------------------
__global__ __launch_bounds__(256) void ln2_kernel(
    const float* __restrict__ xh, const float* __restrict__ g,
    const float* __restrict__ b, _Float16* __restrict__ xn2) {
  __shared__ float tile[16 * 385];
  int bi = blockIdx.x / kTPI;
  int t0 = (blockIdx.x % kTPI) * 16;
  for (int idx = threadIdx.x; idx < 16 * kC; idx += 256) {
    int tok = idx / kC, c = idx - tok * kC;   // lanes coalesced along c
    int hw = t0 + tok;
    tile[tok * 385 + c] = (hw < kHW) ? xh[((size_t)bi * kHW + hw) * kC + c] : 0.f;
  }
  __syncthreads();
  int tok = threadIdx.x >> 4, j = threadIdx.x & 15;
  float sum = 0.f, sq = 0.f;
  for (int c = j; c < kC; c += 16) {
    float v = tile[tok * 385 + c];
    sum += v; sq += v * v;
  }
  for (int d = 8; d >= 1; d >>= 1) {
    sum += __shfl_xor(sum, d);
    sq  += __shfl_xor(sq, d);
  }
  float mean = sum * (1.f / kC);
  float inv  = rsqrtf(sq * (1.f / kC) - mean * mean + 1e-5f);
  int hw = t0 + tok;
  if (hw < kHW) {
    int h = hw / kW, w = hw % kW;
    int n = (bi * kNwinH + (h >> 2)) * kNwinW + (w >> 4);
    int l = ((h & 3) << 4) | (w & 15);
    size_t wb = ((size_t)n * 64 + l) * kC;
    for (int c = j; c < kC; c += 16)
      xn2[wb + c] = (_Float16)((tile[tok * 385 + c] - mean) * inv * g[c] + b[c]);
  }
}

// ---------------------------------------------------------------------------
// 6) Final residual add + NHWC -> NCHW transpose into d_out.
// ---------------------------------------------------------------------------
__global__ __launch_bounds__(256) void final_kernel(
    const float* __restrict__ xh, const float* __restrict__ m,
    float* __restrict__ out) {
  __shared__ float tile[16 * 385];
  int bi = blockIdx.x / kTPI;
  int t0 = (blockIdx.x % kTPI) * 16;
  for (int idx = threadIdx.x; idx < 16 * kC; idx += 256) {
    int tok = idx / kC, c = idx - tok * kC;
    int hw = t0 + tok;
    float v = 0.f;
    if (hw < kHW) {
      int h = hw / kW, w = hw % kW;
      int n = (bi * kNwinH + (h >> 2)) * kNwinW + (w >> 4);
      int l = ((h & 3) << 4) | (w & 15);
      v = xh[((size_t)bi * kHW + hw) * kC + c] + m[((size_t)n * 64 + l) * kC + c];
    }
    tile[tok * 385 + c] = v;
  }
  __syncthreads();
  for (int idx = threadIdx.x; idx < 16 * kC; idx += 256) {
    int tok = idx & 15, c = idx >> 4;       // lanes coalesced along hw
    int hw = t0 + tok;
    if (hw < kHW) out[((size_t)bi * kC + c) * kHW + hw] = tile[tok * 385 + c];
  }
}

// ---------------------------------------------------------------------------
// Host launcher
// ---------------------------------------------------------------------------
extern "C" void kernel_launch(void* const* d_in, const int* in_sizes, int n_in,
                              void* d_out, int out_size, void* d_ws,
                              size_t ws_size, hipStream_t stream) {
  (void)in_sizes; (void)n_in; (void)out_size; (void)ws_size;
  const float* x       = (const float*)d_in[0];
  const float* qkv_w   = (const float*)d_in[1];
  const float* qkv_bn  = (const float*)d_in[2];
  const float* proj_w  = (const float*)d_in[3];
  const float* proj_bn = (const float*)d_in[4];
  const float* n1_g    = (const float*)d_in[5];
  const float* n1_b    = (const float*)d_in[6];
  const float* n2_g    = (const float*)d_in[7];
  const float* n2_b    = (const float*)d_in[8];
  const float* mlp1_w  = (const float*)d_in[9];
  const float* mlp1_bn = (const float*)d_in[10];
  const float* mlp2_w  = (const float*)d_in[11];
  const float* mlp2_bn = (const float*)d_in[12];
  float* out = (float*)d_out;

  char* ws = (char*)d_ws;
  size_t off = 0;
  auto alloc = [&](size_t bytes) -> void* {
    void* p = ws + off;
    off += (bytes + 255) & ~(size_t)255;
    return p;
  };

  _Float16* qkvW  = (_Float16*)alloc((size_t)1152 * 384 * 2);
  float*    qkvB  = (float*)alloc(1152 * 4);
  _Float16* projW = (_Float16*)alloc((size_t)384 * 384 * 2);
  float*    projB = (float*)alloc(384 * 4);
  _Float16* m1W   = (_Float16*)alloc((size_t)768 * 384 * 2);
  float*    m1B   = (float*)alloc(768 * 4);
  _Float16* m2W   = (_Float16*)alloc((size_t)384 * 768 * 2);
  float*    m2B   = (float*)alloc(384 * 4);

  float*    xh   = (float*)alloc((size_t)kB * kHW * kC * 4);
  _Float16* winX = (_Float16*)alloc((size_t)kNT * kC * 2);
  _Float16* Qb   = (_Float16*)alloc((size_t)kNT * kC * 2);
  _Float16* Kb   = (_Float16*)alloc((size_t)kNT * kC * 2);
  _Float16* Vb   = (_Float16*)alloc((size_t)kNT * kC * 2);
  _Float16* oWin = (_Float16*)alloc((size_t)kNT * kC * 2);
  _Float16* xn2  = (_Float16*)alloc((size_t)kNT * kC * 2);
  _Float16* hWin = (_Float16*)alloc((size_t)kNT * 2 * kC * 2);
  float*    mWin = (float*)alloc((size_t)kNT * kC * 4);

  // 1) fold BN into weights (f16)
  fold_bn_kernel<<<(1152 * 384 + 255) / 256, 256, 0, stream>>>(qkv_w, qkv_bn, 1152, 384, qkvW, qkvB);
  fold_bn_kernel<<<(384 * 384 + 255) / 256, 256, 0, stream>>>(proj_w, proj_bn, 384, 384, projW, projB);
  fold_bn_kernel<<<(768 * 384 + 255) / 256, 256, 0, stream>>>(mlp1_w, mlp1_bn, 768, 384, m1W, m1B);
  fold_bn_kernel<<<(384 * 768 + 255) / 256, 256, 0, stream>>>(mlp2_w, mlp2_bn, 384, 768, m2W, m2B);

  // 2) LN1 + transpose + window partition (pads zeroed first)
  hipMemsetAsync(winX, 0, (size_t)kNT * kC * 2, stream);
  ln1_window_kernel<<<kB * kTPI, 256, 0, stream>>>(x, n1_g, n1_b, xh, winX);

  // 3) QKV GEMM -> head-major Q/K/V
  win_gemm_kernel<384, 1152, 0, false><<<kNWin, 256, 0, stream>>>(
      winX, qkvW, qkvB, nullptr, nullptr, Qb, Kb, Vb);

  // 4) windowed multi-head attention
  attn_kernel<<<kNWin * kNH, 128, 0, stream>>>(Qb, Kb, Vb, oWin);

  // 5) proj GEMM with fused crop + residual add into xh
  win_gemm_kernel<384, 384, 1, false><<<kNWin, 256, 0, stream>>>(
      oWin, projW, projB, nullptr, xh, nullptr, nullptr, nullptr);

  // 6) LN2 -> window layout
  ln2_kernel<<<kB * kTPI, 256, 0, stream>>>(xh, n2_g, n2_b, xn2);

  // 7) MLP1 (SiLU fused), MLP2
  win_gemm_kernel<384, 768, 2, true><<<kNWin, 256, 0, stream>>>(
      xn2, m1W, m1B, hWin, nullptr, nullptr, nullptr, nullptr);
  win_gemm_kernel<768, 384, 3, false><<<kNWin, 256, 0, stream>>>(
      hWin, m2W, m2B, nullptr, mWin, nullptr, nullptr, nullptr);

  // 8) residual add + transpose back to NCHW
  final_kernel<<<kB * kTPI, 256, 0, stream>>>(xh, mWin, out);
}
